// ElectronicEmbedding_26121991094371
// MI455X (gfx1250) — matrix operand order repaired and checked by reading, not compiled, not run on hardware
//
#include <hip/hip_runtime.h>
#include <hip/hip_bf16.h>

#define FD 256

typedef __attribute__((ext_vector_type(16))) __bf16 v16bf;
typedef __attribute__((ext_vector_type(8)))  float  v8f;
typedef __attribute__((ext_vector_type(4)))  unsigned int v4u;
typedef __attribute__((ext_vector_type(4)))  unsigned int u32x4;
typedef __attribute__((ext_vector_type(8)))  int i32x8;
typedef __attribute__((ext_vector_type(4)))  int i32x4;

#if defined(__has_builtin)
#if __has_builtin(__builtin_amdgcn_tensor_load_to_lds)
#define HAVE_TDM 1
#endif
#endif

__device__ __forceinline__ unsigned short f2bf(float x) {
  __bf16 b = (__bf16)x;
  return __builtin_bit_cast(unsigned short, b);
}
__device__ __forceinline__ float swishf(float x, float al, float be) {
  // alpha * x * sigmoid(beta * x)
  return al * x * (1.0f / (1.0f + __expf(-be * x)));
}
__device__ __forceinline__ v16bf ld_frag(const unsigned short* p0, const unsigned short* p1) {
  union { v4u u[2]; v16bf v; } r;
  r.u[0] = *(const v4u*)p0;
  r.u[1] = *(const v4u*)p1;
  return r.v;
}

// ---------------------------------------------------------------------------
// Kernel A0: WkWq[2][256] = Wk^T @ Wq  ;  Wkb[2] = Wk^T @ bq
// ---------------------------------------------------------------------------
__global__ void prep_kernel(const float* __restrict__ Wq, const float* __restrict__ bq,
                            const float* __restrict__ Wk,
                            float* __restrict__ wkwq, float* __restrict__ wkb) {
  int i = blockIdx.x;    // 0..1
  int j = threadIdx.x;   // 0..255
  float s = 0.0f;
  for (int f = 0; f < FD; ++f) s += Wk[f * 2 + i] * Wq[f * FD + j];
  wkwq[i * FD + j] = s;
  if (j == 0) {
    float t = 0.0f;
    for (int f = 0; f < FD; ++f) t += Wk[f * 2 + i] * bq[f];
    wkb[i] = t;
  }
}

// ---------------------------------------------------------------------------
// Kernel A1: per-molecule kq_mol, v_mol, kb
// ---------------------------------------------------------------------------
__global__ void mol_kernel(const float* __restrict__ E, const float* __restrict__ Wv,
                           const float* __restrict__ wkwq, const float* __restrict__ wkb,
                           float* __restrict__ kq_mol, float* __restrict__ v_mol,
                           float* __restrict__ kb_mol) {
  int m = blockIdx.x;
  int j = threadIdx.x;
  float e  = E[m];
  float e0 = e > 0.0f ? e : 0.0f;
  float e1 = -e > 0.0f ? -e : 0.0f;
  float c0 = e0 / fmaxf(e0, 1.0f);   // e / clip(e, 1.0)
  float c1 = e1 / fmaxf(e1, 1.0f);
  kq_mol[(size_t)m * FD + j] = c0 * wkwq[j] + c1 * wkwq[FD + j];
  v_mol [(size_t)m * FD + j] = e0 * Wv[j * 2 + 0] + e1 * Wv[j * 2 + 1];
  if (j == 0) kb_mol[m] = c0 * wkb[0] + c1 * wkb[1];
}

// ---------------------------------------------------------------------------
// Kernel Wcvt: pack 5 weight matrices (W1_0,W2_0,W1_1,W2_1,Wout) as bf16
// ---------------------------------------------------------------------------
__global__ void cvt_kernel(const float* __restrict__ w1, const float* __restrict__ w2,
                           const float* __restrict__ wout, unsigned short* __restrict__ wpack) {
  int idx = blockIdx.x * 256 + threadIdx.x;   // 5*65536 total
  int slot = idx >> 16, off = idx & 65535;
  const float* src;
  switch (slot) {
    case 0:  src = w1;          break;
    case 1:  src = w2;          break;
    case 2:  src = w1 + 65536;  break;
    case 3:  src = w2 + 65536;  break;
    default: src = wout;        break;
  }
  wpack[idx] = f2bf(src[off]);
}

// ---------------------------------------------------------------------------
// Phase 1: a[n] = softplus((x[n]·kq_mol[mol] + kb[mol]) / 16); per-block partials
// ---------------------------------------------------------------------------
__global__ void dot_kernel(const float* __restrict__ x, const int* __restrict__ idx,
                           const float* __restrict__ kq, const float* __restrict__ kb,
                           float* __restrict__ a_arr, float* __restrict__ partial, int nAtoms) {
  __shared__ float sred[8];
  int tid = threadIdx.x, wv = tid >> 5, lane = tid & 31;
  int n = blockIdx.x * 8 + wv;
  float myA = 0.0f;
  if (n < nAtoms) {
    const float4* xr = (const float4*)(x + (size_t)n * FD);
    int m = idx[n];
    const float4* kr = (const float4*)(kq + (size_t)m * FD);
    float4 xa = xr[lane], xb = xr[lane + 32];
    float4 ka = kr[lane], kc = kr[lane + 32];
    float d = xa.x * ka.x + xa.y * ka.y + xa.z * ka.z + xa.w * ka.w
            + xb.x * kc.x + xb.y * kc.y + xb.z * kc.z + xb.w * kc.w;
    #pragma unroll
    for (int off = 16; off > 0; off >>= 1) d += __shfl_xor(d, off);
    if (lane == 0) {
      float dd = (d + kb[m]) * 0.0625f;             // /sqrt(256)
      float av = dd > 20.0f ? dd : log1pf(__expf(dd));
      a_arr[n] = av;
      myA = av;
    }
  }
  if (lane == 0) sred[wv] = myA;
  __syncthreads();
  if (tid == 0) {
    float s = 0.0f;
    #pragma unroll
    for (int i = 0; i < 8; ++i) s += sred[i];
    partial[blockIdx.x] = s;
  }
}

__global__ void reduce_kernel(const float* __restrict__ partial, int npart,
                              float* __restrict__ ssum) {
  __shared__ float sd[256];
  float s = 0.0f;
  for (int i = threadIdx.x; i < npart; i += 256) s += partial[i];
  sd[threadIdx.x] = s;
  __syncthreads();
  for (int off = 128; off > 0; off >>= 1) {
    if (threadIdx.x < off) sd[threadIdx.x] += sd[threadIdx.x + off];
    __syncthreads();
  }
  if (threadIdx.x == 0) ssum[0] = sd[0];
}

// ---------------------------------------------------------------------------
// Phase 2: fused residual MLP with bf16 WMMA + TDM weight staging.
// Block = 256 threads (8 waves); each wave owns a 16-atom tile (128 atoms/block).
// ---------------------------------------------------------------------------
#define WSTRIDE 272   // bf16 elems per wbuf row (256 + 16 pad -> 544B rows)
#define TSTRIDE 264   // bf16 elems per tbuf row (256 + 8 pad  -> 528B rows)
#define SMEM_USHORTS (256 * WSTRIDE + 8 * 2 * 16 * TSTRIDE)

__global__ void __launch_bounds__(256, 1)
mlp_kernel(const int* __restrict__ idx, const float* __restrict__ a_arr,
           const float* __restrict__ ssum, const float* __restrict__ v_mol,
           const unsigned short* __restrict__ wpack,
           const float* __restrict__ ra1, const float* __restrict__ rb1,
           const float* __restrict__ ra2, const float* __restrict__ rb2,
           const float* __restrict__ oa,  const float* __restrict__ ob,
           float* __restrict__ out, int nAtoms) {
  extern __shared__ unsigned short sm[];
  unsigned short* wbuf = sm;                       // staged weight matrix (padded rows)
  unsigned short* tb   = sm + 256 * WSTRIDE;       // per-wave activation buffers

  const int tid  = threadIdx.x;
  const int wv   = tid >> 5;
  const int lane = tid & 31;
  const int nlo  = lane & 15;
  const int hx   = lane >> 4;
  unsigned short* t0 = tb + wv * (2 * 16 * TSTRIDE);
  unsigned short* t1 = t0 + 16 * TSTRIDE;
  const int base = blockIdx.x * 128 + wv * 16;

  // --- weight staging: TDM DMA (one instruction) with HW row padding ---
  auto stage = [&](int slot) {
#if defined(HAVE_TDM)
    if (wv == 0) {
      unsigned long long ga = (unsigned long long)(uintptr_t)(wpack + (size_t)slot * 65536);
      unsigned int ldsa = (unsigned int)(uintptr_t)wbuf;   // LDS byte offset (low 32 bits)
      u32x4 g0;
      g0.x = 1u;                                            // count=1, user mode
      g0.y = ldsa;                                          // lds_addr
      g0.z = (unsigned int)(ga & 0xffffffffu);              // global_addr[31:0]
      g0.w = (unsigned int)((ga >> 32) & 0x01ffffffu) | (2u << 30);  // addr[56:32], type=2
      i32x8 g1;
      // data_size=1 (2B), pad_enable, pad_interval=6 (128 DW = 512B), pad_amount=7 (8 DW = 32B)
      g1[0] = (int)((1u << 16) | (1u << 20) | (6u << 22) | (7u << 25));
      g1[1] = (int)(256u << 16);   // tensor_dim0 = 256 (low 16 at bits 63:48)
      g1[2] = (int)(256u << 16);   // tensor_dim1 = 256 (low 16 at bits 95:80)
      g1[3] = (int)(256u << 16);   // tile_dim0 = 256 (bits 127:112)
      g1[4] = 256;                 // tile_dim1 = 256 (bits 143:128), tile_dim2 = 0
      g1[5] = 256;                 // tensor_dim0_stride = 256 (bits 191:160)
      g1[6] = 0;
      g1[7] = 0;
      i32x4 gz4 = {0, 0, 0, 0};
      i32x8 gz8 = {0, 0, 0, 0, 0, 0, 0, 0};
      __builtin_amdgcn_tensor_load_to_lds(g0, g1, gz4, gz4, gz8, 0);
      __builtin_amdgcn_s_wait_tensorcnt(0);
    }
#else
    const v4u* s = (const v4u*)(wpack + (size_t)slot * 65536 + tid * FD);
    v4u* d = (v4u*)(wbuf + tid * WSTRIDE);
    #pragma unroll
    for (int q = 0; q < 32; ++q) d[q] = s[q];
#endif
  };

  // --- build h fragments: h = (a[n]/(S+eps)) * v_mol[mol(n)] ---
  v8f hv[16];
  {
    float inv = 1.0f / (ssum[0] + 1e-8f);
    float coef[8]; const float* vrow[8];
    #pragma unroll
    for (int i = 0; i < 8; ++i) {
      int r  = base + i + 8 * hx;
      int rc = r < nAtoms ? r : nAtoms - 1;
      coef[i] = a_arr[rc] * inv;
      vrow[i] = v_mol + (size_t)idx[rc] * FD;
    }
    #pragma unroll
    for (int nb = 0; nb < 16; ++nb) {
      int col = nb * 16 + nlo;
      #pragma unroll
      for (int i = 0; i < 8; ++i) hv[nb][i] = coef[i] * vrow[i][col];
    }
  }

  // batched per-lane swish parameter preload (hides global latency under WMMA)
  auto ldpar = [&](const float* pa, const float* pb, float* alv, float* bev) {
    #pragma unroll
    for (int nb = 0; nb < 16; ++nb) {
      int col = nb * 16 + nlo;
      alv[nb] = pa[col];
      bev[nb] = pb[col];
    }
  };

  // write t0 = swish(h, pa, pb) as bf16 (C-layout rows: M = i + 8*hx)
  auto write_t_from_h = [&](const float* pa, const float* pb) {
    float alv[16], bev[16];
    ldpar(pa, pb, alv, bev);
    #pragma unroll
    for (int nb = 0; nb < 16; ++nb) {
      int col = nb * 16 + nlo;
      #pragma unroll
      for (int i = 0; i < 8; ++i)
        t0[(i + 8 * hx) * TSTRIDE + col] = f2bf(swishf(hv[nb][i], alv[nb], bev[nb]));
    }
  };

  v16bf afrag[8];
  // A fragment (16x32 bf16): row = lane&15; elems 0..7 -> K = kc*32 + hx*8,
  // elems 8..15 -> K = kc*32 + 16 + hx*8
  auto load_A_all = [&](const unsigned short* tsrc) {
    #pragma unroll
    for (int kc = 0; kc < 8; ++kc) {
      const unsigned short* p = tsrc + nlo * TSTRIDE + kc * 32 + hx * 8;
      afrag[kc] = ld_frag(p, p + 16);
    }
  };
  // B fragment (32x16 bf16): col = lane&15; elems 0..7 -> K = kc*32 + hx*8, elems 8..15 -> K = kc*32 + 16 + hx*8
  auto gemm_nb = [&](int nb) -> v8f {
    v8f c = {0.f, 0.f, 0.f, 0.f, 0.f, 0.f, 0.f, 0.f};
    #pragma unroll
    for (int kc = 0; kc < 8; ++kc) {
      const unsigned short* p = wbuf + (nb * 16 + nlo) * WSTRIDE + kc * 32 + hx * 8;
      v16bf b = ld_frag(p, p + 16);
      c = __builtin_amdgcn_wmma_f32_16x16x32_bf16(false, afrag[kc], false, b,
                                                  (short)0, c, false, false);
    }
    return c;
  };

  // GEMM producing y: t1 = swish(y, pa, pb)
  auto gemm_y = [&](const float* pa, const float* pb) {
    float alv[16], bev[16];
    ldpar(pa, pb, alv, bev);
    load_A_all(t0);
    #pragma unroll
    for (int nb = 0; nb < 16; ++nb) {
      v8f c = gemm_nb(nb);
      int col = nb * 16 + nlo;
      #pragma unroll
      for (int i = 0; i < 8; ++i)
        t1[(i + 8 * hx) * TSTRIDE + col] = f2bf(swishf(c[i], alv[nb], bev[nb]));
    }
  };
  // GEMM closing a residual block: h += z; t0 = swish(h, next params)
  auto gemm_h = [&](const float* npa, const float* npb) {
    float alv[16], bev[16];
    ldpar(npa, npb, alv, bev);
    load_A_all(t1);
    #pragma unroll
    for (int nb = 0; nb < 16; ++nb) {
      v8f c = gemm_nb(nb);
      hv[nb] += c;
      int col = nb * 16 + nlo;
      #pragma unroll
      for (int i = 0; i < 8; ++i)
        t0[(i + 8 * hx) * TSTRIDE + col] = f2bf(swishf(hv[nb][i], alv[nb], bev[nb]));
    }
  };
  // final GEMM: out = swish(h) @ Wout^T
  auto gemm_out = [&]() {
    load_A_all(t0);
    #pragma unroll
    for (int nb = 0; nb < 16; ++nb) {
      v8f c = gemm_nb(nb);
      int col = nb * 16 + nlo;
      #pragma unroll
      for (int i = 0; i < 8; ++i) {
        int r = base + i + 8 * hx;
        if (r < nAtoms) out[(size_t)r * FD + col] = c[i];
      }
    }
  };

  stage(0);
  write_t_from_h(ra1, rb1);            // layer 0: a1[0], b1[0]
  __syncthreads();
  gemm_y(ra2, rb2);                    // y0; t1 = swish(y0, a2[0], b2[0])
  __syncthreads(); stage(1); __syncthreads();
  gemm_h(ra1 + FD, rb1 + FD);          // h += z0; t0 = swish(h, a1[1], b1[1])
  __syncthreads(); stage(2); __syncthreads();
  gemm_y(ra2 + FD, rb2 + FD);          // y1; t1 = swish(y1, a2[1], b2[1])
  __syncthreads(); stage(3); __syncthreads();
  gemm_h(oa, ob);                      // h += z1; t0 = swish(h, out_a, out_b)
  __syncthreads(); stage(4); __syncthreads();
  gemm_out();
}

// ---------------------------------------------------------------------------
extern "C" void kernel_launch(void* const* d_in, const int* in_sizes, int n_in,
                              void* d_out, int out_size, void* d_ws, size_t ws_size,
                              hipStream_t stream) {
  const float* x    = (const float*)d_in[0];
  const float* E    = (const float*)d_in[1];
  const int*   idx  = (const int*)  d_in[2];
  const float* Wq   = (const float*)d_in[3];
  const float* bq   = (const float*)d_in[4];
  const float* Wk   = (const float*)d_in[5];
  const float* Wv   = (const float*)d_in[6];
  const float* rW1  = (const float*)d_in[7];
  const float* rW2  = (const float*)d_in[8];
  const float* ra1  = (const float*)d_in[9];
  const float* rb1  = (const float*)d_in[10];
  const float* ra2  = (const float*)d_in[11];
  const float* rb2  = (const float*)d_in[12];
  const float* oa   = (const float*)d_in[13];
  const float* ob   = (const float*)d_in[14];
  const float* Wout = (const float*)d_in[15];
  float* out = (float*)d_out;

  const int N = in_sizes[0] / FD;
  const int B = in_sizes[1];

  char* ws = (char*)d_ws;
  size_t off = 0;
  auto alloc = [&](size_t bytes) { size_t o = off; off += (bytes + 255) & ~(size_t)255; return o; };
  float* wkwq   = (float*)(ws + alloc(2 * FD * 4));
  float* wkb    = (float*)(ws + alloc(2 * 4));
  float* kq_mol = (float*)(ws + alloc((size_t)B * FD * 4));
  float* v_mol  = (float*)(ws + alloc((size_t)B * FD * 4));
  float* kb_mol = (float*)(ws + alloc((size_t)B * 4));
  float* a_arr  = (float*)(ws + alloc((size_t)N * 4));
  const int nb1 = (N + 7) / 8;
  float* partial = (float*)(ws + alloc((size_t)nb1 * 4));
  float* ssum    = (float*)(ws + alloc(4));
  unsigned short* wpack = (unsigned short*)(ws + alloc((size_t)5 * 65536 * 2));

  prep_kernel<<<2, 256, 0, stream>>>(Wq, bq, Wk, wkwq, wkb);
  mol_kernel<<<B, 256, 0, stream>>>(E, Wv, wkwq, wkb, kq_mol, v_mol, kb_mol);
  cvt_kernel<<<(5 * 65536) / 256, 256, 0, stream>>>(rW1, rW2, Wout, wpack);
  dot_kernel<<<nb1, 256, 0, stream>>>(x, idx, kq_mol, kb_mol, a_arr, partial, N);
  reduce_kernel<<<1, 256, 0, stream>>>(partial, nb1, ssum);

  const int nb2 = (N + 127) / 128;
  const size_t smem = (size_t)SMEM_USHORTS * sizeof(unsigned short);  // 274432 B
  mlp_kernel<<<nb2, 256, smem, stream>>>(idx, a_arr, ssum, v_mol, wpack,
                                         ra1, rb1, ra2, rb2, oa, ob, out, N);
}
